// DVectorNet_309237646053
// MI455X (gfx1250) — matrix operand order repaired and checked by reading, not compiled, not run on hardware
//
#include <hip/hip_runtime.h>

// ---------------------------------------------------------------------------
// DVectorNet on MI455X (gfx1250): LSTM(40->64, T=300) + gather + MLP head.
// v3: same as v2 (register-resident B fragments, TDM double-buffered x-tile
//     streaming) but using the 6-argument tensor_load_to_lds builtin this
//     toolchain declares: (u32x4, i32x8, i32x4, i32x4, i32x8, i32 cpol).
// ---------------------------------------------------------------------------

typedef __attribute__((ext_vector_type(16))) _Float16 v16h;
typedef __attribute__((ext_vector_type(8)))  float    v8f;
typedef unsigned int u32x4 __attribute__((ext_vector_type(4)));
typedef int          i32x4 __attribute__((ext_vector_type(4)));
typedef int          i32x8 __attribute__((ext_vector_type(8)));

#define B_   2048
#define T_   300
#define D_   40
#define H_   64
#define DH_  104   // D + H (K of the gate GEMM)
#define KP_  128   // K padded to multiple of 32 (A zero-padded there)
#define G4_  256   // 4*H
#define D1_  128
#define OUT_ 512

// workspace layout (bytes); total = 724992 B
#define WS_FINALH 0                          // 2048*64*4  = 524288
#define WS_WT     524288                     // 256*104*2  =  53248
#define WS_W1T    (524288 + 53248)           // 128*64*2   =  16384
#define WS_W2T    (524288 + 53248 + 16384)   // 512*128*2  = 131072

#if __has_builtin(__builtin_amdgcn_tensor_load_to_lds) && \
    __has_builtin(__builtin_amdgcn_s_wait_tensorcnt)
#define USE_TDM 1
#else
#define USE_TDM 0
#endif

union Frag { uint4 q[2]; v16h v; };

// generic -> LDS(addrspace 3) byte offset for TDM descriptors
__device__ static inline unsigned lds_byte_off(const void* p) {
  return (unsigned)(unsigned long long)
      (const __attribute__((address_space(3))) char*)p;
}

#if USE_TDM
// Issue a TDM load of a [16 x 40] f32 tile (rows stride T*D floats apart)
// from gptr into LDS at byte offset ldsoff.  2-D descriptor per ISA 8.3/8.4.
__device__ static inline void tdm_load_x(const float* gptr, unsigned ldsoff) {
  unsigned long long ga = (unsigned long long)gptr;
  u32x4 g0;
  g0.x = 1u;                                   // count=1, user descriptor
  g0.y = ldsoff;                               // lds_addr [63:32]
  g0.z = (unsigned)(ga & 0xffffffffull);       // global_addr [95:64]
  g0.w = (unsigned)((ga >> 32) & 0x01ffffffull) | (2u << 30); // addr hi | type=2
  i32x8 g1;
  g1[0] = 0x00020000;                          // data_size=4B, mask=0
  g1[1] = (int)(D_ << 16);                     // tensor_dim0 = 40 (bits 79:48)
  g1[2] = (int)(16 << 16);                     // tensor_dim1 = 16 (bits 111:80)
  g1[3] = (int)(D_ << 16);                     // tile_dim0 = 40 (bits 127:112)
  g1[4] = 16;                                  // tile_dim1 = 16, tile_dim2 = 0
  g1[5] = T_ * D_;                             // tensor_dim0_stride = 12000
  g1[6] = 0;
  g1[7] = 0;
  i32x4 z4 = {0, 0, 0, 0};                     // 2-D: groups 2/3 unused
  i32x8 z8 = {0, 0, 0, 0, 0, 0, 0, 0};
  __builtin_amdgcn_tensor_load_to_lds(g0, g1, z4, z4, z8, 0);
}
#endif

// ---------------------------------------------------------------------------
// Kernel 0: transpose + f32->f16 convert all weight matrices into workspace.
// ---------------------------------------------------------------------------
__global__ void prep_weights(const float* __restrict__ lstm_k,
                             const float* __restrict__ W1,
                             const float* __restrict__ W2,
                             _Float16* __restrict__ wt,
                             _Float16* __restrict__ w1t,
                             _Float16* __restrict__ w2t) {
  int idx = blockIdx.x * blockDim.x + threadIdx.x;
  int stride = gridDim.x * blockDim.x;
  for (int i = idx; i < G4_ * DH_; i += stride) {
    int n = i / DH_, k = i % DH_;
    wt[i] = (_Float16)lstm_k[k * G4_ + n];
  }
  for (int i = idx; i < D1_ * H_; i += stride) {
    int n = i >> 6, k = i & 63;
    w1t[i] = (_Float16)W1[k * D1_ + n];
  }
  for (int i = idx; i < OUT_ * D1_; i += stride) {
    int n = i >> 7, k = i & 127;
    w2t[i] = (_Float16)W2[k * OUT_ + n];
  }
}

// ---------------------------------------------------------------------------
// Kernel 1: fused LSTM recurrence. 128 blocks x 256 threads (8 waves).
// Block owns batch rows [16*wg, 16*wg+16). Wave wv -> N-tiles {2wv, 2wv+1};
// B fragments live in VGPRs for all 300 steps.
// ---------------------------------------------------------------------------
__global__ __launch_bounds__(256) void lstm_fused(
    const float* __restrict__ X, const int* __restrict__ seqlen,
    const float* __restrict__ bias, const _Float16* __restrict__ wt_g,
    float* __restrict__ finalH) {
  __shared__ alignas(16) float    sX[2][16 * D_];   // 5 KB TDM double buffer
  __shared__ alignas(16) _Float16 sA[16 * KP_];     // 4 KB [x|h|0] per row
  __shared__ alignas(16) _Float16 sZ[16 * G4_];     // 8 KB gate pre-acts

  const int tid  = threadIdx.x;
  const int wg   = blockIdx.x;
  const int lane = tid & 31;
  const int wv   = tid >> 5;
  const int m    = lane & 15;
  const int hi   = lane >> 4;
  const int nt0  = wv * 2;

  // ---- resident B fragments: 2 tiles x 4 K-chunks, loaded once from L2.
  // Reads past a row / past the array hit the adjacent w1t bytes; those K
  // positions multiply A's zero padding, contributing exactly 0.
  Frag bfr[2][4];
#pragma unroll
  for (int j = 0; j < 2; ++j)
#pragma unroll
    for (int kc = 0; kc < 4; ++kc) {
      const _Float16* wr = wt_g + ((nt0 + j) * 16 + m) * DH_ + kc * 32 + hi * 16;
      bfr[j][kc].q[0] = *(const uint4*)(wr);
      bfr[j][kc].q[1] = *(const uint4*)(wr + 8);
    }

  // zero h-region and K pad: halves [40,128) of every row (pad stays 0)
  for (int i = tid; i < 16 * (KP_ - D_); i += 256) {
    int r = i / (KP_ - D_), k = D_ + i % (KP_ - D_);
    sA[r * KP_ + k] = (_Float16)0.f;
  }

  // elementwise mapping: thread -> (row er, units eu..eu+3), fixed all steps
  const int er   = tid >> 4;
  const int eu   = (tid & 15) * 4;
  const int brow = wg * 16 + er;
  const int myseq = seqlen[brow] - 1;           // seq in [1,T] -> always hit
  float c[4]  = {0.f, 0.f, 0.f, 0.f};
  float hf[4] = {0.f, 0.f, 0.f, 0.f};
  float4 bi = *(const float4*)(bias + eu);
  float4 bj = *(const float4*)(bias + H_ + eu);
  float4 bf = *(const float4*)(bias + 2 * H_ + eu);
  float4 bo = *(const float4*)(bias + 3 * H_ + eu);

#if USE_TDM
  const unsigned xbase = lds_byte_off(&sX[0][0]);
  if (wv == 0)                                   // pre-issue step 0 tile
    tdm_load_x(X + (size_t)wg * 16 * T_ * D_, xbase);
#endif

  for (int t = 0; t < T_; ++t) {
#if USE_TDM
    const int buf = t & 1;
    if (wv == 0) __builtin_amdgcn_s_wait_tensorcnt(0);   // buf(t) landed
    __syncthreads();                                     // publish buf(t)
    for (int i = tid; i < 16 * D_; i += 256) {           // f32 -> f16 into A
      int r = i / D_, k = i % D_;
      sA[r * KP_ + k] = (_Float16)sX[buf][i];
    }
    if (wv == 0 && (t + 1) < T_)                         // prefetch t+1 tile
      tdm_load_x(X + ((size_t)wg * 16 * T_ + (t + 1)) * D_,
                 xbase + (unsigned)(buf ^ 1) * (16 * D_ * 4));
    __syncthreads();                                     // sA ready
#else
    for (int i = tid; i < 16 * D_; i += 256) {
      int r = i / D_, k = i % D_;
      size_t gi = ((size_t)(wg * 16 + r) * T_ + t) * D_ + k;
      sA[r * KP_ + k] = (_Float16)X[gi];
    }
    if (tid < 16 && (t + 1) < T_)
      __builtin_prefetch(&X[((size_t)(wg * 16 + tid) * T_ + t + 1) * D_], 0, 3);
    __syncthreads();
#endif

    // ---- z tiles: A fragments from LDS, resident B fragments in VGPRs
    Frag a[4];
#pragma unroll
    for (int kc = 0; kc < 4; ++kc) {
      int ka = kc * 32 + hi * 16;      // A: lanes>=16 start at K+16
      a[kc].q[0] = *(const uint4*)(sA + m * KP_ + ka);
      a[kc].q[1] = *(const uint4*)(sA + m * KP_ + ka + 8);
    }
    v8f acc0 = {}; v8f acc1 = {};
#pragma unroll
    for (int kc = 0; kc < 4; ++kc) {
      acc0 = __builtin_amdgcn_wmma_f32_16x16x32_f16(false, a[kc].v, false,
                 bfr[0][kc].v, (short)0, acc0, false, false);
      acc1 = __builtin_amdgcn_wmma_f32_16x16x32_f16(false, a[kc].v, false,
                 bfr[1][kc].v, (short)0, acc1, false, false);
    }
#pragma unroll
    for (int v = 0; v < 8; ++v) {
      int zr = v + hi * 8;             // C layout: M = v + 8*(lane>=16)
      sZ[zr * G4_ + nt0 * 16 + m]       = (_Float16)acc0[v];
      sZ[zr * G4_ + (nt0 + 1) * 16 + m] = (_Float16)acc1[v];
    }
    __syncthreads();

    // ---- cell update for (er, eu..eu+3)
    {
      const _Float16* zr = sZ + er * G4_;
      union { _Float16 h[4]; uint2 u; } hp;
#pragma unroll
      for (int k = 0; k < 4; ++k) {
        float iv = (float)zr[eu + k]          + ((float*)&bi)[k];
        float jv = (float)zr[H_ + eu + k]     + ((float*)&bj)[k];
        float fv = (float)zr[2 * H_ + eu + k] + ((float*)&bf)[k] + 1.0f;
        float ov = (float)zr[3 * H_ + eu + k] + ((float*)&bo)[k];
        float si = 1.f / (1.f + __expf(-iv));
        float sf = 1.f / (1.f + __expf(-fv));
        float so = 1.f / (1.f + __expf(-ov));
        float nc = c[k] * sf + si * tanhf(jv);
        float nh = tanhf(nc) * so;
        c[k] = nc;
        hp.h[k] = (_Float16)nh;
        if (t == myseq) hf[k] = nh;
      }
      *(uint2*)(sA + er * KP_ + D_ + eu) = hp.u;  // h feeds next step
    }
  }

  float4 o4 = {hf[0], hf[1], hf[2], hf[3]};
  *(float4*)(finalH + (size_t)brow * H_ + eu) = o4;
}

// ---------------------------------------------------------------------------
// Kernel 2: head. 128 blocks x 256 threads; block owns 16 batch rows.
// ---------------------------------------------------------------------------
__global__ __launch_bounds__(256) void head_kernel(
    const float* __restrict__ finalH,
    const _Float16* __restrict__ w1t, const float* __restrict__ b1,
    const float* __restrict__ gamma, const float* __restrict__ beta,
    const float* __restrict__ mmean, const float* __restrict__ mvar,
    const _Float16* __restrict__ w2t, const float* __restrict__ b2,
    float* __restrict__ out) {
  __shared__ alignas(16) _Float16 sA1[16 * H_];
  __shared__ alignas(16) _Float16 sA2[16 * D1_];
  __shared__ alignas(16) float sLogit[16 * OUT_];
  __shared__ float sRed[16 * 16];
  __shared__ float sRowMax[16];
  __shared__ float sRowSum[16];

  const int tid  = threadIdx.x;
  const int wg   = blockIdx.x;
  const int lane = tid & 31;
  const int wv   = tid >> 5;
  const int m    = lane & 15;
  const int hi   = lane >> 4;

  for (int i = tid; i < 16 * H_; i += 256)
    sA1[i] = (_Float16)finalH[(size_t)wg * 16 * H_ + i];
  __syncthreads();

  // ---- dense1: wave wv -> N-tile wv, K=64
  {
    v8f acc = {};
#pragma unroll
    for (int kc = 0; kc < 2; ++kc) {
      Frag a, b;
      int ka = kc * 32 + hi * 16;
      a.q[0] = *(const uint4*)(sA1 + m * H_ + ka);
      a.q[1] = *(const uint4*)(sA1 + m * H_ + ka + 8);
      int kb = kc * 32 + hi * 16;
      const _Float16* wr = w1t + (wv * 16 + m) * H_ + kb;
      b.q[0] = *(const uint4*)(wr);
      b.q[1] = *(const uint4*)(wr + 8);
      acc = __builtin_amdgcn_wmma_f32_16x16x32_f16(false, a.v, false, b.v,
                                                   (short)0, acc, false, false);
    }
    int n = wv * 16 + m;
    float bb = b1[n], g = gamma[n], be = beta[n], mu = mmean[n];
    float inv = rsqrtf(mvar[n] + 1e-3f);
#pragma unroll
    for (int v = 0; v < 8; ++v) {
      int r = v + hi * 8;
      float x = acc[v] + bb;
      x = x > 0.f ? x : 0.f;                    // ReLU
      x = g * (x - mu) * inv + be;              // inference BN
      sA2[r * D1_ + n] = (_Float16)x;
    }
  }
  __syncthreads();

  // ---- dense2: wave wv -> N-tiles {wv, wv+8, wv+16, wv+24}, K=128
  {
    v8f acc[4] = {{}, {}, {}, {}};
#pragma unroll
    for (int kc = 0; kc < 4; ++kc) {
      Frag a;
      int ka = kc * 32 + hi * 16;
      a.q[0] = *(const uint4*)(sA2 + m * D1_ + ka);
      a.q[1] = *(const uint4*)(sA2 + m * D1_ + ka + 8);
      int kb = kc * 32 + hi * 16;
#pragma unroll
      for (int j = 0; j < 4; ++j) {
        Frag b;
        const _Float16* wr = w2t + ((wv + 8 * j) * 16 + m) * D1_ + kb;
        b.q[0] = *(const uint4*)(wr);
        b.q[1] = *(const uint4*)(wr + 8);
        acc[j] = __builtin_amdgcn_wmma_f32_16x16x32_f16(false, a.v, false, b.v,
                                                        (short)0, acc[j], false, false);
      }
    }
#pragma unroll
    for (int j = 0; j < 4; ++j) {
      int n = (wv + 8 * j) * 16 + m;
      float bb = b2[n];
#pragma unroll
      for (int v = 0; v < 8; ++v) {
        int r = v + hi * 8;
        sLogit[r * OUT_ + n] = acc[j][v] + bb;
      }
    }
  }
  __syncthreads();

  // ---- softmax: 16 threads per row, 32 cols each
  const int r  = tid >> 4;
  const int jj = tid & 15;
  float mx = -1e30f;
  for (int n = jj * 32; n < jj * 32 + 32; ++n)
    mx = fmaxf(mx, sLogit[r * OUT_ + n]);
  sRed[r * 16 + jj] = mx;
  __syncthreads();
  if (jj == 0) {
    float v = sRed[r * 16];
    for (int k = 1; k < 16; ++k) v = fmaxf(v, sRed[r * 16 + k]);
    sRowMax[r] = v;
  }
  __syncthreads();
  const float rmax = sRowMax[r];
  float sum = 0.f;
  for (int n = jj * 32; n < jj * 32 + 32; ++n) {
    float p = __expf(sLogit[r * OUT_ + n] - rmax);
    sLogit[r * OUT_ + n] = p;
    sum += p;
  }
  sRed[r * 16 + jj] = sum;
  __syncthreads();
  if (jj == 0) {
    float v = 0.f;
    for (int k = 0; k < 16; ++k) v += sRed[r * 16 + k];
    sRowSum[r] = v;
  }
  __syncthreads();
  const float inv = 1.0f / sRowSum[r];
  for (int n = jj * 32; n < jj * 32 + 32; ++n)
    out[(size_t)(wg * 16 + r) * OUT_ + n] = sLogit[r * OUT_ + n] * inv;
}

// ---------------------------------------------------------------------------
extern "C" void kernel_launch(void* const* d_in, const int* in_sizes, int n_in,
                              void* d_out, int out_size, void* d_ws, size_t ws_size,
                              hipStream_t stream) {
  (void)in_sizes; (void)n_in; (void)out_size; (void)ws_size;
  const float* X      = (const float*)d_in[0];
  const int*   seq    = (const int*)d_in[1];
  const float* lstm_k = (const float*)d_in[2];
  const float* lstm_b = (const float*)d_in[3];
  const float* W1     = (const float*)d_in[4];
  const float* b1     = (const float*)d_in[5];
  const float* gamma  = (const float*)d_in[6];
  const float* beta   = (const float*)d_in[7];
  const float* mmean  = (const float*)d_in[8];
  const float* mvar   = (const float*)d_in[9];
  const float* W2     = (const float*)d_in[10];
  const float* b2     = (const float*)d_in[11];
  float* out = (float*)d_out;

  char* ws = (char*)d_ws;                       // needs ~708 KB of scratch
  float*    finalH = (float*)(ws + WS_FINALH);
  _Float16* wt     = (_Float16*)(ws + WS_WT);
  _Float16* w1t    = (_Float16*)(ws + WS_W1T);
  _Float16* w2t    = (_Float16*)(ws + WS_W2T);

  prep_weights<<<64, 256, 0, stream>>>(lstm_k, W1, W2, wt, w1t, w2t);
  lstm_fused<<<B_ / 16, 256, 0, stream>>>(X, seq, lstm_b, wt, finalH);
  head_kernel<<<B_ / 16, 256, 0, stream>>>(finalH, w1t, b1, gamma, beta,
                                           mmean, mvar, w2t, b2, out);
}
